// SlowFastSSMM_9869834846610
// MI455X (gfx1250) — matrix operand order repaired and checked by reading, not compiled, not run on hardware
//
#include <hip/hip_runtime.h>
#include <hip/hip_bf16.h>

// ---------------------------------------------------------------------------
// SlowFastSSMM forward for MI455X (gfx1250), wave32 + WMMA bf16.
//
// Problem constants (from reference):
//   B=2, T=64000, L_F=L_S=32, D_F=D_S=16, H=32, GH=64, N_GRU=4
//   nf = ns = 3999, BS = B*ns = 7998, delta = 1  (A_flat == A_s)
// ---------------------------------------------------------------------------

#define T_LEN   64000
#define B_SZ    2
#define NS      3999
#define NF      3999
#define BSW     7998      // B * ns
#define GH_     64
#define LS_     32
#define H_      32
#define LOG2E_  1.4426950408889634f

typedef __attribute__((ext_vector_type(16))) __bf16 bf16x16;
typedef __attribute__((ext_vector_type(8)))  float  f32x8;

union Frag16 { bf16x16 v; uint4 q[2]; };

__device__ __forceinline__ f32x8 wmma_bf16(bf16x16 a, bf16x16 b, f32x8 c) {
  return __builtin_amdgcn_wmma_f32_16x16x32_bf16(
      /*neg_a=*/false, a, /*neg_b=*/false, b,
      /*c_mod=*/(short)0, c, /*reuse_a=*/false, /*reuse_b=*/false);
}

// Hardware-approximate transcendentals (v_exp_f32 = 2^x, v_rcp_f32).
// Saturation: x->+inf => exp2->0 => sigmoid->1 ; x->-inf => rcp(inf)=0.
__device__ __forceinline__ float sigmoidf_(float v) {
  return __builtin_amdgcn_rcpf(1.0f + __builtin_amdgcn_exp2f(-v * LOG2E_));
}
__device__ __forceinline__ float tanhf_(float v) {
  return 1.0f - 2.0f * __builtin_amdgcn_rcpf(
                          1.0f + __builtin_amdgcn_exp2f(2.0f * LOG2E_ * v));
}

// A-fragment (16x32 bf16, M rows x K cols) from LDS, row-major with rowStride.
// lane<16 : rows m=lane,    K = k0+[0..7] and k0+16+[0..7]
// lane>=16: rows m=lane-16, K = k0+8+[0..7] and k0+24+[0..7]
__device__ __forceinline__ bf16x16 load_a_frag(const __bf16* base, int rowStride,
                                               int k0, int m, int half) {
  Frag16 f;
  const __bf16* p = base + m * rowStride + k0 + 8 * half;
  f.q[0] = *(const uint4*)(p);
  f.q[1] = *(const uint4*)(p + 16);
  return f.v;
}

// B-fragment (32x16 bf16, K x N) from transposed-weight LDS wt[n*64 + k].
// lane holds column n = ncol; elements i<8: k = 32*kb + 8*half + i, i>=8: k = 32*kb + 16 + 8*half + (i-8).
__device__ __forceinline__ bf16x16 load_b_frag(const __bf16* wt, int ncol,
                                               int kb, int half) {
  Frag16 f;
  const __bf16* p = wt + ncol * GH_ + 32 * kb + 8 * half;
  f.q[0] = *(const uint4*)(p);
  f.q[1] = *(const uint4*)(p + 16);
  return f.v;
}

// ---------------------------------------------------------------------------
// Kernel 1: slow path. One workgroup = 4 waves = 128 threads handles a tile of
// 16 windows through slow_in -> 4 GRU layers -> slow_out -> s2s, writing
// A_s (sigmoid) and g_s per window to workspace.
// Wave w owns hidden columns [16w, 16w+16): gate tiles r/z/n at N-offsets
// {16w, 64+16w, 128+16w} so its hn slice equals its hc slice (kept in C regs).
// Single barrier per recurrent step: hc broadcast buffer is double-buffered
// on t-parity, and the relu(hn) -> sH[t] store is deferred to step t+1.
// ---------------------------------------------------------------------------
__global__ __launch_bounds__(128) void slow_gru_kernel(
    const float* __restrict__ x,
    const float* __restrict__ siW,  const float* __restrict__ sib,
    const float* __restrict__ Wih,  const float* __restrict__ Whh,
    const float* __restrict__ bih,  const float* __restrict__ bhh,
    const float* __restrict__ soW,  const float* __restrict__ sob,
    const float* __restrict__ s2sW, const float* __restrict__ s2sb,
    float* __restrict__ A_ws, float* __restrict__ g_ws)
{
  extern __shared__ char smem[];
  __bf16* sWi = (__bf16*)(smem);            // Wih^T  [192][64]      24576 B
  __bf16* sWh = (__bf16*)(smem + 24576);    // Whh^T  [192][64]      24576 B
  __bf16* sH  = (__bf16*)(smem + 49152);    // h_seq  [16][32][64]   65536 B
  __bf16* sHC = (__bf16*)(smem + 114688);   // hc x2  [2][16][64]     4096 B

  const int tid  = threadIdx.x;
  const int w    = tid >> 5;
  const int lane = tid & 31;
  const int half = lane >> 4;
  const int ln   = lane & 15;
  const int wg   = blockIdx.x;
  const int j0   = 16 * w;                  // wave's hidden-column base

  // ---- h0 = relu(x_win * slow_in_W + slow_in_b), stored bf16 ----
  for (int idx = tid; idx < 16 * LS_ * GH_; idx += 128) {
    int win = idx >> 11;
    int t   = (idx >> 6) & 31;
    int j   = idx & 63;
    int g   = wg * 16 + win; if (g > BSW - 1) g = BSW - 1;
    int b   = g / NS, s0 = g - b * NS;
    float xv = x[b * T_LEN + s0 * 16 + t];
    sH[idx] = (__bf16)fmaxf(xv * siW[j] + sib[j], 0.0f);
  }

  f32x8 hc;
  for (int l = 0; l < 4; ++l) {
    __syncthreads();
    // ---- stage layer weights, transposed + bf16, into LDS ----
    const float* wi = Wih + l * GH_ * 192;
    const float* wh = Whh + l * GH_ * 192;
    for (int idx = tid; idx < GH_ * 192; idx += 128) {
      int k = idx / 192, n = idx - k * 192;
      sWi[n * GH_ + k] = (__bf16)wi[idx];
      sWh[n * GH_ + k] = (__bf16)wh[idx];
    }
    for (int idx = tid; idx < 16 * GH_; idx += 128) sHC[idx] = (__bf16)0.0f;

    const float bsr  = bih[l * 192 +       j0 + ln] + bhh[l * 192 +       j0 + ln];
    const float bsz  = bih[l * 192 +  64 + j0 + ln] + bhh[l * 192 +  64 + j0 + ln];
    const float bin_ = bih[l * 192 + 128 + j0 + ln];
    const float bhn  = bhh[l * 192 + 128 + j0 + ln];
#pragma unroll
    for (int r = 0; r < 8; ++r) hc[r] = 0.0f;
    __syncthreads();

    // ---- preload loop-invariant B fragments (r/z/n tiles x K-blocks) ----
    bf16x16 bwi[6], bwh[6];
#pragma unroll
    for (int tl = 0; tl < 3; ++tl)
#pragma unroll
      for (int kb = 0; kb < 2; ++kb) {
        int nc = 64 * tl + j0 + ln;
        bwi[tl * 2 + kb] = load_b_frag(sWi, nc, kb, half);
        bwh[tl * 2 + kb] = load_b_frag(sWh, nc, kb, half);
      }

    // ---- 32 recurrent steps, one barrier each ----
    __bf16 hrelu[8];
    for (int t = 0; t < LS_; ++t) {
      // deferred store of step t-1 relu output (disjoint from step-t reads)
      if (t > 0) {
#pragma unroll
        for (int r = 0; r < 8; ++r) {
          int m = r + 8 * half;
          sH[(m * LS_ + (t - 1)) * GH_ + j0 + ln] = hrelu[r];
        }
      }
      const __bf16* sHCr = sHC + (t & 1) * (16 * GH_);
      __bf16*       sHCw = sHC + ((t + 1) & 1) * (16 * GH_);

      const __bf16* hx = sH + t * GH_;  // row stride 32*64 over windows
      bf16x16 ax0 = load_a_frag(hx,   LS_ * GH_, 0,  ln, half);
      bf16x16 ax1 = load_a_frag(hx,   LS_ * GH_, 32, ln, half);
      bf16x16 ah0 = load_a_frag(sHCr, GH_,       0,  ln, half);
      bf16x16 ah1 = load_a_frag(sHCr, GH_,       32, ln, half);

      f32x8 accx[3], acch[3];
#pragma unroll
      for (int tl = 0; tl < 3; ++tl) {
#pragma unroll
        for (int r = 0; r < 8; ++r) { accx[tl][r] = 0.0f; acch[tl][r] = 0.0f; }
        accx[tl] = wmma_bf16(ax0, bwi[tl * 2 + 0], accx[tl]);
        accx[tl] = wmma_bf16(ax1, bwi[tl * 2 + 1], accx[tl]);
        acch[tl] = wmma_bf16(ah0, bwh[tl * 2 + 0], acch[tl]);
        acch[tl] = wmma_bf16(ah1, bwh[tl * 2 + 1], acch[tl]);
      }

#pragma unroll
      for (int r = 0; r < 8; ++r) {
        float rr = sigmoidf_(accx[0][r] + acch[0][r] + bsr);
        float zz = sigmoidf_(accx[1][r] + acch[1][r] + bsz);
        float nn = tanhf_(accx[2][r] + bin_ + rr * (acch[2][r] + bhn));
        float hn = nn + zz * (hc[r] - nn);
        hc[r] = hn;                              // pre-relu recurrent state
        int m = r + 8 * half;
        sHCw[m * GH_ + j0 + ln] = (__bf16)hn;
        hrelu[r] = (__bf16)fmaxf(hn, 0.0f);
      }
      __syncthreads();  // step-t writes visible; step-t reads all done
    }
    // flush deferred t=31 relu output
#pragma unroll
    for (int r = 0; r < 8; ++r) {
      int m = r + 8 * half;
      sH[(m * LS_ + 31) * GH_ + j0 + ln] = hrelu[r];
    }
  }

  // ---- slow_out: slow_feat = relu(h[:, -1, :]) @ slow_out_W + b ----
  __syncthreads();
  for (int idx = tid; idx < GH_ * GH_; idx += 128) {
    int k = idx >> 6, n = idx & 63;
    sWi[n * GH_ + k] = (__bf16)soW[idx];    // slow_out_W^T
    sWh[n * GH_ + k] = (__bf16)s2sW[idx];   // s2s_W^T
  }
  __syncthreads();

  const __bf16* hl = sH + 31 * GH_;
  bf16x16 a0 = load_a_frag(hl, LS_ * GH_, 0,  ln, half);
  bf16x16 a1 = load_a_frag(hl, LS_ * GH_, 32, ln, half);
  bf16x16 b0 = load_b_frag(sWi, j0 + ln, 0, half);
  bf16x16 b1 = load_b_frag(sWi, j0 + ln, 1, half);
  f32x8 sf;
#pragma unroll
  for (int r = 0; r < 8; ++r) sf[r] = 0.0f;
  sf = wmma_bf16(a0, b0, sf);
  sf = wmma_bf16(a1, b1, sf);
  const float sob_l = sob[j0 + ln];
  __syncthreads();
#pragma unroll
  for (int r = 0; r < 8; ++r) {
    int m = r + 8 * half;
    sHC[m * GH_ + j0 + ln] = (__bf16)(sf[r] + sob_l);  // slow_feat, bf16
  }
  __syncthreads();

  // ---- eps = slow_feat @ s2s_W + b; A_s = sigmoid(eps[:H]), g_s = eps[H:] ----
  bf16x16 c0 = load_a_frag(sHC, GH_, 0,  ln, half);
  bf16x16 c1 = load_a_frag(sHC, GH_, 32, ln, half);
  bf16x16 d0 = load_b_frag(sWh, j0 + ln, 0, half);
  bf16x16 d1 = load_b_frag(sWh, j0 + ln, 1, half);
  f32x8 ep;
#pragma unroll
  for (int r = 0; r < 8; ++r) ep[r] = 0.0f;
  ep = wmma_bf16(c0, d0, ep);
  ep = wmma_bf16(c1, d1, ep);
  const float s2b = s2sb[j0 + ln];
#pragma unroll
  for (int r = 0; r < 8; ++r) {
    int m = r + 8 * half;
    int g = wg * 16 + m;
    if (g < BSW) {
      float v = ep[r] + s2b;
      int col = j0 + ln;
      if (col < H_) A_ws[g * H_ + col]        = sigmoidf_(v);
      else          g_ws[g * H_ + (col - H_)] = v;
    }
  }
}

// ---------------------------------------------------------------------------
// Kernel 2: fast diagonal SSM. One wave32 per frame, lane = channel h (H=32).
// ---------------------------------------------------------------------------
__global__ __launch_bounds__(256) void fast_ssm_kernel(
    const float* __restrict__ x,
    const float* __restrict__ finW, const float* __restrict__ finb,
    const float* __restrict__ foutW, const float* __restrict__ foutb,
    const float* __restrict__ A_ws, const float* __restrict__ g_ws,
    float* __restrict__ s_ws)
{
  int f = blockIdx.x * 8 + (threadIdx.x >> 5);
  int lane = threadIdx.x & 31;
  if (f >= B_SZ * NF) return;
  int b = f / NF, i = f - b * NF;

  float xv = x[b * T_LEN + i * 16 + lane];     // frame sample for step 'lane'
  float A  = A_ws[f * H_ + lane];
  float g  = g_ws[f * H_ + lane];
  float fw = finW[lane], fb = finb[lane];
  float ow = foutW[lane], ob = foutb[0];

  float hc = 0.0f, sres = 0.0f;
  for (int t = 0; t < LS_; ++t) {
    float xt = __shfl(xv, t, 32);
    hc = A * hc + (xt * fw + fb) * g;
    float p = hc * ow;
    p += __shfl_xor(p, 16, 32);
    p += __shfl_xor(p,  8, 32);
    p += __shfl_xor(p,  4, 32);
    p += __shfl_xor(p,  2, 32);
    p += __shfl_xor(p,  1, 32);
    if (lane == t) sres = p + ob;              // lane t keeps s_hat[t]
  }
  s_ws[f * LS_ + lane] = sres;                 // coalesced 32-wide store
}

// ---------------------------------------------------------------------------
// Kernel 3: overlap-add gather (each sample gets <=2 frame contributions).
// ---------------------------------------------------------------------------
__global__ __launch_bounds__(256) void overlap_add_kernel(
    const float* __restrict__ s_ws, float* __restrict__ out)
{
  int idx = blockIdx.x * 256 + threadIdx.x;
  if (idx >= B_SZ * T_LEN) return;
  int b = idx / T_LEN, t = idx - b * T_LEN;
  int fh = t >> 4, j = t & 15;
  float acc = 0.0f;
  if (fh < NF)  acc += s_ws[(b * NF + fh) * LS_ + j];
  if (fh >= 1)  acc += s_ws[(b * NF + fh - 1) * LS_ + j + 16];
  out[idx] = acc;
}

// ---------------------------------------------------------------------------
extern "C" void kernel_launch(void* const* d_in, const int* in_sizes, int n_in,
                              void* d_out, int out_size, void* d_ws, size_t ws_size,
                              hipStream_t stream) {
  const float* x     = (const float*)d_in[0];
  const float* siW   = (const float*)d_in[1];
  const float* sib   = (const float*)d_in[2];
  const float* gWih  = (const float*)d_in[3];
  const float* gWhh  = (const float*)d_in[4];
  const float* gbih  = (const float*)d_in[5];
  const float* gbhh  = (const float*)d_in[6];
  const float* soW   = (const float*)d_in[7];
  const float* sob   = (const float*)d_in[8];
  const float* s2sW  = (const float*)d_in[9];
  const float* s2sb  = (const float*)d_in[10];
  const float* finW  = (const float*)d_in[11];
  const float* finb  = (const float*)d_in[12];
  const float* foutW = (const float*)d_in[13];
  const float* foutb = (const float*)d_in[14];
  float* out = (float*)d_out;

  // workspace carve-up (floats): A_s | g_s | s_hat, padded to 8000 windows
  float* ws   = (float*)d_ws;
  float* A_ws = ws;                 // 8000*32
  float* g_ws = ws + 256000;        // 8000*32
  float* s_ws = ws + 512000;        // 8000*32

  const int nWG  = (BSW + 15) / 16;                 // 500
  const size_t smem = 24576 + 24576 + 65536 + 4096; // 118784 B

  slow_gru_kernel<<<nWG, 128, smem, stream>>>(
      x, siW, sib, gWih, gWhh, gbih, gbhh, soW, sob, s2sW, s2sb, A_ws, g_ws);

  fast_ssm_kernel<<<(B_SZ * NF + 7) / 8, 256, 0, stream>>>(
      x, finW, finb, foutW, foutb, A_ws, g_ws, s_ws);

  overlap_add_kernel<<<(B_SZ * T_LEN + 255) / 256, 256, 0, stream>>>(s_ws, out);
}
